// VoteQuery_84705345012393
// MI455X (gfx1250) — compile-verified, matrix operand order checked by reading, not run on hardware
//
#include <hip/hip_runtime.h>
#include <hip/hip_bf16.h>

// Problem constants (match reference)
#define DV    256
#define NPTS  16384
#define BATCH 8
#define NQV   256
#define NSAMP 16
#define NB    64          // feature columns per block in the FFN kernel
#define RADI  0.3f
#define EPSB  1e-5f

#if defined(__has_builtin)
#if __has_builtin(__builtin_amdgcn_global_load_async_to_lds_b128)
#define ASYNC_LDS_OK 1
#endif
#endif
#ifndef ASYNC_LDS_OK
#define ASYNC_LDS_OK 0
#endif

typedef __attribute__((ext_vector_type(16))) __bf16 v16bf;
typedef __attribute__((ext_vector_type(8)))  float  v8f;
typedef __attribute__((ext_vector_type(4)))  int    v4i;
typedef __attribute__((address_space(1)))    v4i    gv4i;   // global 16B chunk
typedef __attribute__((address_space(3)))    v4i    lv4i;   // LDS 16B chunk

__device__ __forceinline__ v8f wmma_bf16(v16bf a, v16bf b, v8f c) {
  // D = A(16x32) * B(32x16) + C, f32 accumulate
  return __builtin_amdgcn_wmma_f32_16x16x32_bf16(false, a, false, b, (short)0, c, false, false);
}

// A-fragment from pre-converted, pre-padded bf16 weights [rows x LDW] row-major.
template<int LDW>
__device__ __forceinline__ v16bf load_a_bf16(const __bf16* __restrict__ W,
                                             int mt, int kb, int lane) {
  int m  = mt * 16 + (lane & 15);
  int k0 = kb * 32 + (lane >> 4) * 8;
  const __bf16* p = W + (size_t)m * LDW + k0;
  v16bf a;
#pragma unroll
  for (int e = 0; e < 8; ++e) a[e]     = p[e];
#pragma unroll
  for (int e = 0; e < 8; ++e) a[8 + e] = p[16 + e];
  return a;
}

// B-fragment from LDS, activation X stored [n][k] bf16 with row stride LDK.
template<int LDK>
__device__ __forceinline__ v16bf load_b_frag(const __bf16* __restrict__ Xs, int nt, int kb, int lane) {
  int n  = nt * 16 + (lane & 15);
  int k0 = kb * 32 + (lane >> 4) * 8;
  const __bf16* p = Xs + (size_t)n * LDK + k0;
  v16bf r;
#pragma unroll
  for (int e = 0; e < 8; ++e) r[e]     = p[e];
#pragma unroll
  for (int e = 0; e < 8; ++e) r[8 + e] = p[16 + e];
  return r;
}

// ------------------------------------------------------------------
// Prep kernels: convert fp32 weights to zero-padded bf16 (once per call)
// ------------------------------------------------------------------
__global__ __launch_bounds__(256)
void k_cvt_pad(const float* __restrict__ src, __bf16* __restrict__ dst,
               int rows_src, int cols_src, int rows_dst, int cols_dst) {
  int i = blockIdx.x * 256 + threadIdx.x;
  int total = rows_dst * cols_dst;
  if (i >= total) return;
  int r = i / cols_dst, c = i - r * cols_dst;
  float v = (r < rows_src && c < cols_src) ? src[(size_t)r * cols_src + c] : 0.0f;
  dst[i] = (__bf16)v;
}

__global__ __launch_bounds__(256)
void k_pad_bias(const float* __restrict__ src, float* __restrict__ dst, int n_src, int n_dst) {
  int i = blockIdx.x * 256 + threadIdx.x;
  if (i < n_dst) dst[i] = (i < n_src) ? src[i] : 0.0f;
}

// ------------------------------------------------------------------
// Kernel A: fused FFN_vote over NB=64 columns per block (dynamic LDS ~102KB).
// ------------------------------------------------------------------
__device__ __forceinline__ void ffn_layer64(const __bf16* __restrict__ Wb,
                                            const __bf16* __restrict__ Xin,
                                            __bf16* __restrict__ Yout,
                                            const float* __restrict__ sScale,
                                            const float* __restrict__ sShift,
                                            int lane, int wave) {
#pragma unroll
  for (int t = 0; t < 2; ++t) {
    int mt = wave + t * 8;
    v8f acc[4];
#pragma unroll
    for (int i = 0; i < 4; ++i) acc[i] = (v8f){};
#pragma unroll
    for (int kb = 0; kb < 8; ++kb) {
      v16bf a = load_a_bf16<256>(Wb, mt, kb, lane);
#pragma unroll
      for (int nt = 0; nt < 4; ++nt) {
        v16bf bb = load_b_frag<256>(Xin, nt, kb, lane);
        acc[nt] = wmma_bf16(a, bb, acc[nt]);
      }
    }
    int ncol  = lane & 15;
    int mbase = mt * 16 + 8 * (lane >> 4);
#pragma unroll
    for (int nt = 0; nt < 4; ++nt) {
      __bf16* qp = Yout + (size_t)(nt * 16 + ncol) * 256 + mbase;
#pragma unroll
      for (int vv = 0; vv < 8; ++vv) {
        float y = acc[nt][vv] * sScale[mbase + vv] + sShift[mbase + vv];
        qp[vv] = (__bf16)fmaxf(y, 0.0f);
      }
    }
  }
}

__global__ __launch_bounds__(256)
void k_ffn_vote(const float* __restrict__ enc_xyz, const float* __restrict__ enc_feat,
                const __bf16* __restrict__ w1b,
                const float* __restrict__ cb1, const float* __restrict__ g1,
                const float* __restrict__ bb1, const float* __restrict__ m1, const float* __restrict__ v1,
                const __bf16* __restrict__ w2b,
                const float* __restrict__ cb2, const float* __restrict__ g2,
                const float* __restrict__ bb2, const float* __restrict__ m2, const float* __restrict__ v2,
                const __bf16* __restrict__ w3b, const float* __restrict__ cb3p,
                float* __restrict__ out_vote, __bf16* __restrict__ featsN) {
  extern __shared__ __align__(16) char smem[];
  __bf16* sX = (__bf16*)smem;                    // 32 KB  [n][k], ld 256
  char*   yo = smem + (size_t)NB * 256 * 2;
  __bf16* Y1 = (__bf16*)yo;                      // 32 KB bf16 [n][k]
  float*  O  = (float*)yo;                       // 68 KB f32 [n][272]

  __shared__ float sScale[256];
  __shared__ float sShift[256];
  __shared__ float sRed[256];

  int blk = blockIdx.x;
  int b   = blk / (NPTS / NB);
  int n0  = (blk % (NPTS / NB)) * NB;
  int tid = threadIdx.x;
  int lane = tid & 31, wave = tid >> 5;

#if ASYNC_LDS_OK
  // Async DMA the raw fp32 tile [c][n] into the (idle) O region, then
  // transpose/convert LDS->LDS to bf16 [n][k].  Uses ASYNCcnt path.
  {
    float* stage = O;  // 64 KB staging inside the 68 KB union region
#pragma unroll
    for (int it = 0; it < 16; ++it) {
      int e  = it * 256 + tid;
      int c  = e >> 4;          // channel row (256B each => 16 chunks of 16B)
      int ck = e & 15;
      const float* gp = enc_feat + ((size_t)b * DV + c) * NPTS + n0 + ck * 4;
      float* lp = stage + (size_t)c * NB + ck * 4;
      __builtin_amdgcn_global_load_async_to_lds_b128((gv4i*)gp, (lv4i*)lp, 0, 0);
    }
#if __has_builtin(__builtin_amdgcn_s_wait_asynccnt)
    __builtin_amdgcn_s_wait_asynccnt(0);
#else
    asm volatile("s_wait_asynccnt 0x0" ::: "memory");
#endif
    __syncthreads();
#pragma unroll
    for (int it = 0; it < (NB * 256) / 256; ++it) {
      int e = it * 256 + tid;
      int n = e & (NB - 1);
      int c = e >> 6;
      sX[(size_t)n * 256 + c] = (__bf16)stage[(size_t)c * NB + n];
    }
  }
#else
  // Fallback: direct global fp32 -> LDS bf16 transpose
#pragma unroll
  for (int it = 0; it < (NB * 256) / 256; ++it) {
    int e = it * 256 + tid;
    int n = e & (NB - 1);
    int c = e >> 6;
    sX[(size_t)n * 256 + c] = (__bf16)enc_feat[((size_t)b * DV + c) * NPTS + n0 + n];
  }
#endif

  // ---- layer 1 ----
  __syncthreads();
  {
    float sc = g1[tid] * rsqrtf(v1[tid] + EPSB);
    sScale[tid] = sc;
    sShift[tid] = (cb1[tid] - m1[tid]) * sc + bb1[tid];
  }
  __syncthreads();
  ffn_layer64(w1b, sX, Y1, sScale, sShift, lane, wave);

  // ---- layer 2 ----
  __syncthreads();
  {
    float sc = g2[tid] * rsqrtf(v2[tid] + EPSB);
    sScale[tid] = sc;
    sShift[tid] = (cb2[tid] - m2[tid]) * sc + bb2[tid];
  }
  __syncthreads();
  ffn_layer64(w2b, Y1, sX, sScale, sShift, lane, wave);   // Y2 -> sX
  __syncthreads();

  // ---- layer 3: padded [272 x 256] GEMM, fp32 out into O ----
#pragma unroll
  for (int t = 0; t < 3; ++t) {
    int mt;
    if (t < 2) mt = wave + t * 8;
    else { if (wave != 0) break; mt = 16; }
    v8f acc[4];
#pragma unroll
    for (int i = 0; i < 4; ++i) acc[i] = (v8f){};
#pragma unroll
    for (int kb = 0; kb < 8; ++kb) {
      v16bf a = load_a_bf16<256>(w3b, mt, kb, lane);
#pragma unroll
      for (int nt = 0; nt < 4; ++nt) {
        v16bf bb = load_b_frag<256>(sX, nt, kb, lane);
        acc[nt] = wmma_bf16(a, bb, acc[nt]);
      }
    }
    int ncol  = lane & 15;
    int mbase = mt * 16 + 8 * (lane >> 4);
#pragma unroll
    for (int nt = 0; nt < 4; ++nt) {
#pragma unroll
      for (int vv = 0; vv < 8; ++vv) {
        int row = mbase + vv;
        O[(size_t)(nt * 16 + ncol) * 272 + row] = acc[nt][vv] + cb3p[row];
      }
    }
  }
  __syncthreads();

  // ---- epilogue: residual feats + L2 normalize, vote_xyz ----
  int col  = tid & (NB - 1);
  int part = tid >> 6;  // 0..3, 64 channels each
  const float* encf = enc_feat + (size_t)b * DV * NPTS + n0 + col;
  float ss = 0.0f;
#pragma unroll
  for (int j = 0; j < 64; ++j) {
    int c = part * 64 + j;
    float f = encf[(size_t)c * NPTS] + O[(size_t)col * 272 + 3 + c];
    O[(size_t)col * 272 + 3 + c] = f;    // stash residual value in-place
    ss += f * f;
  }
  sRed[part * 64 + col] = ss;
  __syncthreads();
  if (tid < NB) {
    float s = 0.0f;
#pragma unroll
    for (int p = 0; p < 4; ++p) s += sRed[p * 64 + tid];
    sRed[tid] = rsqrtf(s);
  }
  __syncthreads();
  float rn = sRed[col];
#pragma unroll
  for (int j = 0; j < 64; ++j) {
    int c = part * 64 + j;
    featsN[((size_t)b * DV + c) * NPTS + n0 + col] = (__bf16)(O[(size_t)col * 272 + 3 + c] * rn);
  }
  if (tid < NB) {
    size_t base = ((size_t)b * NPTS + n0 + tid) * 3;
#pragma unroll
    for (int d = 0; d < 3; ++d)
      out_vote[base + d] = enc_xyz[base + d] + O[(size_t)tid * 272 + d];
  }
}

// ------------------------------------------------------------------
// Kernel B: furthest-point sampling (256 points) + gather new_xyz from vote_xyz
// ------------------------------------------------------------------
__global__ __launch_bounds__(256)
void k_fps(const float* __restrict__ xyz, const float* __restrict__ vote,
           float* __restrict__ newxyz, float* __restrict__ mind_ws) {
  __shared__ float sBest[8];
  __shared__ int   sBidx[8];
  __shared__ int   sCur;
  __shared__ int   sInds[NQV];

  int b = blockIdx.x;
  int tid = threadIdx.x, lane = tid & 31, wave = tid >> 5;
  const float* X = xyz + (size_t)b * NPTS * 3;
  float* mind = mind_ws + (size_t)b * NPTS;

  for (int s = 0; s < NPTS / 256; ++s) mind[s * 256 + tid] = 1e10f;

  int cur = 0;
  for (int j = 0; j < NQV; ++j) {
    if (tid == 0) sInds[j] = cur;
    float cx = X[(size_t)cur * 3 + 0];
    float cy = X[(size_t)cur * 3 + 1];
    float cz = X[(size_t)cur * 3 + 2];
    float best = -1.0f; int bidx = 0;
    for (int s = 0; s < NPTS / 256; ++s) {
      int i = s * 256 + tid;
      float dx = X[(size_t)i * 3 + 0] - cx;
      float dy = X[(size_t)i * 3 + 1] - cy;
      float dz = X[(size_t)i * 3 + 2] - cz;
      float d  = dx * dx + dy * dy + dz * dz;
      float m  = fminf(mind[i], d);
      mind[i]  = m;
      if (m > best) { best = m; bidx = i; }
    }
#pragma unroll
    for (int off = 16; off > 0; off >>= 1) {
      float ob = __shfl_down(best, off, 32);
      int   oi = __shfl_down(bidx, off, 32);
      if (ob > best || (ob == best && oi < bidx)) { best = ob; bidx = oi; }
    }
    if (lane == 0) { sBest[wave] = best; sBidx[wave] = bidx; }
    __syncthreads();
    if (tid == 0) {
      float bb = sBest[0]; int bi = sBidx[0];
      for (int w = 1; w < 8; ++w)
        if (sBest[w] > bb || (sBest[w] == bb && sBidx[w] < bi)) { bb = sBest[w]; bi = sBidx[w]; }
      sCur = bi;
    }
    __syncthreads();
    cur = sCur;
    __syncthreads();
  }
  int ind = sInds[tid];
  const float* vp = vote + ((size_t)b * NPTS + ind) * 3;
  float* op = newxyz + ((size_t)b * NQV + tid) * 3;
  op[0] = vp[0]; op[1] = vp[1]; op[2] = vp[2];
}

// ------------------------------------------------------------------
// Kernel C: ball query + grouping + 3 SA GEMMs (WMMA bf16) + maxpool
// ------------------------------------------------------------------
template<int LDW, int LDK, int LDO, int KB>
__device__ __forceinline__ void sa_layer(const __bf16* __restrict__ Wb,
                                         const __bf16* __restrict__ Xin,
                                         __bf16* __restrict__ Ybf, float* __restrict__ Yf32,
                                         const float* __restrict__ sScale,
                                         const float* __restrict__ sShift,
                                         int lane, int wave) {
#pragma unroll
  for (int t = 0; t < 2; ++t) {
    int mt = wave + t * 8;
    v8f acc = (v8f){};
#pragma unroll
    for (int kb = 0; kb < KB; ++kb) {
      v16bf a  = load_a_bf16<LDW>(Wb, mt, kb, lane);
      v16bf bb = load_b_frag<LDK>(Xin, 0, kb, lane);
      acc = wmma_bf16(a, bb, acc);
    }
    int sC    = lane & 15;
    int mbase = mt * 16 + 8 * (lane >> 4);
#pragma unroll
    for (int vv = 0; vv < 8; ++vv) {
      float y = fmaxf(acc[vv] * sScale[mbase + vv] + sShift[mbase + vv], 0.0f);
      if (Ybf) Ybf[(size_t)sC * LDO + mbase + vv] = (__bf16)y;
      else     Yf32[(size_t)(mbase + vv) * NSAMP + sC] = y;
    }
  }
}

__global__ __launch_bounds__(256)
void k_query(const float* __restrict__ vote, const float* __restrict__ newxyz,
             const __bf16* __restrict__ featsN,
             const __bf16* __restrict__ s0wb,
             const float* __restrict__ s0g, const float* __restrict__ s0b,
             const float* __restrict__ s0m, const float* __restrict__ s0v,
             const __bf16* __restrict__ s1wb,
             const float* __restrict__ s1g, const float* __restrict__ s1b,
             const float* __restrict__ s1m, const float* __restrict__ s1v,
             const __bf16* __restrict__ s2wb,
             const float* __restrict__ s2g, const float* __restrict__ s2b,
             const float* __restrict__ s2m, const float* __restrict__ s2v,
             float* __restrict__ qf_out) {
  __shared__ int sIdx[NSAMP];
  __shared__ int sWaveCnt[8];
  __shared__ int sWaveHits[8][NSAMP];
  __shared__ __align__(16) __bf16 sG[NSAMP * 288];  // [s][k], K padded to 288
  __shared__ __align__(16) __bf16 sH[NSAMP * 256];  // [s][k]
  __shared__ float sScale[256], sShift[256];
  __shared__ float sO[256 * NSAMP];                 // [c][s]

  int blk = blockIdx.x;
  int b = blk / NQV, q = blk % NQV;
  int tid = threadIdx.x, lane = tid & 31, wave = tid >> 5;

  float nx = newxyz[((size_t)b * NQV + q) * 3 + 0];
  float ny = newxyz[((size_t)b * NQV + q) * 3 + 1];
  float nz = newxyz[((size_t)b * NQV + q) * 3 + 2];

  // ---- ball query: each wave scans 2048 indices, keeps first 16 hits ----
  {
    int cnt = 0;
    int base = wave * (NPTS / 8);
    for (int s0 = 0; s0 < NPTS / 8; s0 += 32) {
      if (cnt >= NSAMP) break;
      int i = base + s0 + lane;
      const float* vp = vote + ((size_t)b * NPTS + i) * 3;
      float dx = vp[0] - nx, dy = vp[1] - ny, dz = vp[2] - nz;
      bool hit = (dx * dx + dy * dy + dz * dz) < (RADI * RADI);
      unsigned mask = (unsigned)__ballot(hit);
      int pos = cnt + __popc(mask & ((1u << lane) - 1u));
      if (hit && pos < NSAMP) sWaveHits[wave][pos] = i;
      cnt += __popc(mask);
    }
    if (lane == 0) sWaveCnt[wave] = (cnt < NSAMP) ? cnt : NSAMP;
  }
  __syncthreads();
  if (tid == 0) {
    int total = 0;
    for (int w = 0; w < 8 && total < NSAMP; ++w) {
      int c2 = sWaveCnt[w];
      for (int e = 0; e < c2 && total < NSAMP; ++e) sIdx[total++] = sWaveHits[w][e];
    }
    int pad = (total > 0) ? sIdx[0] : 0;
    for (; total < NSAMP; ++total) sIdx[total] = pad;
  }
  __syncthreads();

  // ---- build group matrix g [s][288] bf16 ----
  for (int e = tid; e < NSAMP * 288; e += 256) sG[e] = (__bf16)0.0f;
  __syncthreads();
  if (tid < NSAMP) {
    int i = sIdx[tid];
    const float* vp = vote + ((size_t)b * NPTS + i) * 3;
    sG[(size_t)tid * 288 + 0] = (__bf16)((vp[0] - nx) * (1.0f / RADI));
    sG[(size_t)tid * 288 + 1] = (__bf16)((vp[1] - ny) * (1.0f / RADI));
    sG[(size_t)tid * 288 + 2] = (__bf16)((vp[2] - nz) * (1.0f / RADI));
  }
  {
    int c = tid;  // 0..255
    const __bf16* fp = featsN + ((size_t)b * DV + c) * NPTS;
#pragma unroll
    for (int s = 0; s < NSAMP; ++s)
      sG[(size_t)s * 288 + 3 + c] = fp[sIdx[s]];
  }

  // ---- SA layer 0: padded [256 x 288] ----
  __syncthreads();
  { float sc = s0g[tid] * rsqrtf(s0v[tid] + EPSB); sScale[tid] = sc; sShift[tid] = s0b[tid] - s0m[tid] * sc; }
  __syncthreads();
  sa_layer<288, 288, 256, 9>(s0wb, sG, sH, nullptr, sScale, sShift, lane, wave);

  // ---- SA layer 1: [256 x 256] ----
  __syncthreads();
  { float sc = s1g[tid] * rsqrtf(s1v[tid] + EPSB); sScale[tid] = sc; sShift[tid] = s1b[tid] - s1m[tid] * sc; }
  __syncthreads();
  sa_layer<256, 256, 288, 8>(s1wb, sH, sG, nullptr, sScale, sShift, lane, wave);

  // ---- SA layer 2: [256 x 256], fp32 out to sO ----
  __syncthreads();
  { float sc = s2g[tid] * rsqrtf(s2v[tid] + EPSB); sScale[tid] = sc; sShift[tid] = s2b[tid] - s2m[tid] * sc; }
  __syncthreads();
  sa_layer<256, 288, 0, 8>(s2wb, sG, nullptr, sO, sScale, sShift, lane, wave);
  __syncthreads();

  // ---- maxpool over 16 samples ----
  {
    int c = tid;
    float mx = sO[(size_t)c * NSAMP];
#pragma unroll
    for (int s = 1; s < NSAMP; ++s) mx = fmaxf(mx, sO[(size_t)c * NSAMP + s]);
    qf_out[((size_t)b * DV + c) * NQV + q] = mx;
  }
}

// ------------------------------------------------------------------
extern "C" void kernel_launch(void* const* d_in, const int* in_sizes, int n_in,
                              void* d_out, int out_size, void* d_ws, size_t ws_size,
                              hipStream_t stream) {
  (void)in_sizes; (void)n_in; (void)out_size; (void)ws_size;
  const float* enc_xyz  = (const float*)d_in[0];
  const float* enc_feat = (const float*)d_in[1];
  const float* w1  = (const float*)d_in[2];  const float* cb1 = (const float*)d_in[3];
  const float* g1  = (const float*)d_in[4];  const float* bb1 = (const float*)d_in[5];
  const float* m1  = (const float*)d_in[6];  const float* v1  = (const float*)d_in[7];
  const float* w2  = (const float*)d_in[8];  const float* cb2 = (const float*)d_in[9];
  const float* g2  = (const float*)d_in[10]; const float* bb2 = (const float*)d_in[11];
  const float* m2  = (const float*)d_in[12]; const float* v2  = (const float*)d_in[13];
  const float* w3  = (const float*)d_in[14]; const float* cb3 = (const float*)d_in[15];
  const float* s0w = (const float*)d_in[16];
  const float* s0g = (const float*)d_in[17]; const float* s0b = (const float*)d_in[18];
  const float* s0m = (const float*)d_in[19]; const float* s0v = (const float*)d_in[20];
  const float* s1w = (const float*)d_in[21];
  const float* s1g = (const float*)d_in[22]; const float* s1b = (const float*)d_in[23];
  const float* s1m = (const float*)d_in[24]; const float* s1v = (const float*)d_in[25];
  const float* s2w = (const float*)d_in[26];
  const float* s2g = (const float*)d_in[27]; const float* s2b = (const float*)d_in[28];
  const float* s2m = (const float*)d_in[29]; const float* s2v = (const float*)d_in[30];

  float* out        = (float*)d_out;
  float* vote       = out;                                   // [B,N,3]
  float* encxyz_out = vote + (size_t)BATCH * NPTS * 3;       // [B,N,3]
  float* newxyz     = encxyz_out + (size_t)BATCH * NPTS * 3; // [B,NQ,3]
  float* qf         = newxyz + (size_t)BATCH * NQV * 3;      // [B,D,NQ]

  // ---- workspace layout ----
  char* ws = (char*)d_ws;
  size_t off = 0;
  __bf16* featsN = (__bf16*)(ws + off); off += (size_t)BATCH * DV * NPTS * 2;  // 64 MB
  float*  mind   = (float*)(ws + off);  off += (size_t)BATCH * NPTS * 4;       // 512 KB
  __bf16* w1b    = (__bf16*)(ws + off); off += (size_t)256 * 256 * 2;
  __bf16* w2b    = (__bf16*)(ws + off); off += (size_t)256 * 256 * 2;
  __bf16* w3b    = (__bf16*)(ws + off); off += (size_t)272 * 256 * 2;
  __bf16* s0wb   = (__bf16*)(ws + off); off += (size_t)256 * 288 * 2;
  __bf16* s1wb   = (__bf16*)(ws + off); off += (size_t)256 * 256 * 2;
  __bf16* s2wb   = (__bf16*)(ws + off); off += (size_t)256 * 256 * 2;
  float*  cb3p   = (float*)(ws + off);  off += (size_t)272 * 4;

  // ---- weight prep (fp32 -> padded bf16) ----
  int g6 = (256 * 256 + 255) / 256;
  k_cvt_pad<<<g6, 256, 0, stream>>>(w1, w1b, 256, 256, 256, 256);
  k_cvt_pad<<<g6, 256, 0, stream>>>(w2, w2b, 256, 256, 256, 256);
  k_cvt_pad<<<(272 * 256 + 255) / 256, 256, 0, stream>>>(w3, w3b, 259, 256, 272, 256);
  k_cvt_pad<<<(256 * 288 + 255) / 256, 256, 0, stream>>>(s0w, s0wb, 256, 259, 256, 288);
  k_cvt_pad<<<g6, 256, 0, stream>>>(s1w, s1wb, 256, 256, 256, 256);
  k_cvt_pad<<<g6, 256, 0, stream>>>(s2w, s2wb, 256, 256, 256, 256);
  k_pad_bias<<<(272 + 255) / 256, 256, 0, stream>>>(cb3, cb3p, 259, 272);

  // ---- main pipeline ----
  size_t ldsA = (size_t)NB * 256 * 2 + (size_t)272 * NB * 4;  // 32KB + 68KB
  k_ffn_vote<<<BATCH * NPTS / NB, 256, ldsA, stream>>>(
      enc_xyz, enc_feat,
      w1b, cb1, g1, bb1, m1, v1,
      w2b, cb2, g2, bb2, m2, v2,
      w3b, cb3p, vote, featsN);

  (void)hipMemcpyAsync(encxyz_out, enc_xyz, (size_t)BATCH * NPTS * 3 * sizeof(float),
                       hipMemcpyDeviceToDevice, stream);

  k_fps<<<BATCH, 256, 0, stream>>>(enc_xyz, vote, newxyz, mind);

  k_query<<<BATCH * NQV, 256, 0, stream>>>(
      vote, newxyz, featsN,
      s0wb, s0g, s0b, s0m, s0v,
      s1wb, s1g, s1b, s1m, s1v,
      s2wb, s2g, s2b, s2m, s2v,
      qf);
}
